// TransformAttentionModel_85014582657131
// MI455X (gfx1250) — matrix operand order repaired and checked by reading, compiled, not run on hardware
//
#include <hip/hip_runtime.h>
#include <hip/hip_bf16.h>

// ---------------------------------------------------------------------------
// TransformAttentionModel on MI455X (gfx1250, wave32, WMMA bf16 f32-acc,
// async-to-LDS staging).
//
// Pipeline:
//   1) cvt_weights: Wq,Wk,Wv,W1,W2 f32 -> bf16 *transposed* Wt[n][k] (ws)
//   2) gemm512<f32 in, relu, bf16 out>: q = relu(STE_Q@Wq+bq)  -> bufQ
//   3)  "                               k = relu(STE_P@Wk+bk)  -> bufK
//   4)  "                               v = relu(X@Wv+bv)      -> bufV
//   5) attn_kernel: per (b,n,head) wave: scores=qk^T/8 (2 WMMA),
//      softmax over TP via width-16 shuffles, attn@v (4 WMMA) -> bufC,
//      attn probs (f32) -> d_out tail in [K*B, N, TQ, TP] layout
//   6) gemm512<bf16 in, relu, bf16 out>: h   = relu(bufC@W1+b1) -> bufQ
//   7) gemm512<bf16 in, none, f32 out>:  out = h@W2+b2          -> d_out head
//
// CDNA5 paths: v_wmma_f32_16x16x32_bf16 everywhere, W/A tiles staged with
// global_load_async_to_lds_b128 (ASYNCcnt) where the source is bf16,
// global_prefetch for the f32 activation stream.
// ---------------------------------------------------------------------------

typedef __attribute__((ext_vector_type(16))) __bf16          v16bf;
typedef __attribute__((ext_vector_type(8)))  float           v8f;
typedef __attribute__((ext_vector_type(16))) unsigned short  v16u;
typedef __attribute__((ext_vector_type(8)))  unsigned short  v8u;

struct u16x16 { v8u lo, hi; };  // 32B, bit-castable to v16bf

__device__ __forceinline__ unsigned short f32_to_bf16(float f) {
  unsigned u = __builtin_bit_cast(unsigned, f);
  return (unsigned short)((u + 0x7FFFu + ((u >> 16) & 1u)) >> 16);  // RNE
}
__device__ __forceinline__ unsigned pk_bf16x2(float a, float b) {
  return (unsigned)f32_to_bf16(a) | ((unsigned)f32_to_bf16(b) << 16);
}

// --- gfx1250 async copy: global -> LDS, 16B per lane, tracked by ASYNCcnt ---
__device__ __forceinline__ void async_g2l_b128(unsigned lds_off, const void* g) {
  asm volatile("global_load_async_to_lds_b128 %0, %1, off"
               :: "v"(lds_off), "v"(g) : "memory");
}
__device__ __forceinline__ void wait_async0() {
#if __has_builtin(__builtin_amdgcn_s_wait_asynccnt)
  __builtin_amdgcn_s_wait_asynccnt(0);
#else
  asm volatile("s_wait_asynccnt 0x0" ::: "memory");
#endif
}
__device__ __forceinline__ unsigned lds_off32(const void* p) {
  return (unsigned)(uintptr_t)p;  // flat LDS aperture: addr[31:0] = wg offset
}

#define D_MODEL 512
#define N_NODES 1024
#define TT      12          // TQ == TP == 12
#define NHEAD   8
#define D_HEAD  64
#define BATCH   8
#define M_ROWS  (BATCH * TT * N_NODES)   // 98304

// ---------------------------------------------------------------------------
// Weight down-convert + transpose: Wt[n][k] = bf16(W[k][n]), 5 matrices
// ---------------------------------------------------------------------------
__global__ void __launch_bounds__(256)
cvt_weights(const float* __restrict__ Wq, const float* __restrict__ Wk,
            const float* __restrict__ Wv, const float* __restrict__ W1,
            const float* __restrict__ W2, unsigned short* __restrict__ out) {
  const int SZ = D_MODEL * D_MODEL;
  const int i = blockIdx.x * blockDim.x + threadIdx.x;   // grid covers SZ
  const int k = i >> 9, n = i & (D_MODEL - 1);
  const int t = n * D_MODEL + k;                         // transposed index
  out[t]          = f32_to_bf16(Wq[i]);
  out[SZ + t]     = f32_to_bf16(Wk[i]);
  out[2 * SZ + t] = f32_to_bf16(Wv[i]);
  out[3 * SZ + t] = f32_to_bf16(W1[i]);
  out[4 * SZ + t] = f32_to_bf16(W2[i]);
}

// ---------------------------------------------------------------------------
// GEMM: Out[M,512] = act( A[M,512] @ W[512,512] + bias ),  W given transposed
// Block 256 threads (8 waves), tile 128x64, BK=32, K-loop of 16.
// Wave grid 4(M) x 2(N); each wave: 2x2 tiles of v_wmma_f32_16x16x32_bf16.
// bf16 tiles staged via global_load_async_to_lds_b128.
// ---------------------------------------------------------------------------
template <bool A_IS_F32, bool RELU, bool OUT_F32>
__global__ void __launch_bounds__(256)
gemm512(const void* __restrict__ Aptr, const unsigned short* __restrict__ Wt,
        const float* __restrict__ bias, void* __restrict__ Out) {
  constexpr int BM = 128, BN = 64, BK = 32, KD = D_MODEL, LDP = BK + 8;
  __shared__ unsigned short lA[BM][LDP];    // A tile, row-major (bf16)
  __shared__ unsigned short lBt[BN][LDP];   // W tile already transposed: [n][k]

  const int tid   = threadIdx.x;
  const int wave  = tid >> 5;
  const int lane  = tid & 31;
  const int waveM = wave >> 1;        // 0..3
  const int waveN = wave & 1;         // 0..1
  const int lc    = lane & 15;
  const int hiL   = lane >> 4;
  const int bm    = blockIdx.x * BM;
  const int bn    = blockIdx.y * BN;
  const int akb   = hiL * 16;         // A-frag K base (contiguous lane-half split)

  // staging geometry (fixed across K loop)
  const int arow = tid >> 1;                   // 0..127
  const int acol = (tid & 1) * 16;             // 0 / 16
  const int wn   = tid >> 2;                   // 0..63
  const int wk   = (tid & 3) * 8;              // 0,8,16,24
  const unsigned ldsA = lds_off32(&lA[arow][acol]);
  const unsigned ldsW = lds_off32(&lBt[wn][wk]);
  const unsigned short* gW = Wt + (size_t)(bn + wn) * KD + wk;

  v8f acc[2][2] = {};

  for (int k0 = 0; k0 < KD; k0 += BK) {
    // --- stage W tile (64x32 of Wt) : one async b128 per thread
    async_g2l_b128(ldsW, gW + k0);

    // --- stage A tile (128x32)
    if (A_IS_F32) {
      const float* A = (const float*)Aptr + (size_t)(bm + arow) * KD + k0 + acol;
      float4 f0 = ((const float4*)A)[0];
      float4 f1 = ((const float4*)A)[1];
      float4 f2 = ((const float4*)A)[2];
      float4 f3 = ((const float4*)A)[3];
      if (k0 + BK < KD) __builtin_prefetch(A + BK, 0, 0);  // global_prefetch
      ((uint4*)&lA[arow][acol])[0] =
          make_uint4(pk_bf16x2(f0.x, f0.y), pk_bf16x2(f0.z, f0.w),
                     pk_bf16x2(f1.x, f1.y), pk_bf16x2(f1.z, f1.w));
      ((uint4*)&lA[arow][acol + 8])[0] =
          make_uint4(pk_bf16x2(f2.x, f2.y), pk_bf16x2(f2.z, f2.w),
                     pk_bf16x2(f3.x, f3.y), pk_bf16x2(f3.z, f3.w));
    } else {
      const unsigned short* A =
          (const unsigned short*)Aptr + (size_t)(bm + arow) * KD + k0 + acol;
      async_g2l_b128(ldsA, A);
      async_g2l_b128(ldsA + 16, A + 8);
    }

    wait_async0();
    __syncthreads();

    #pragma unroll
    for (int sm = 0; sm < 2; ++sm) {
      const int ar = waveM * 32 + sm * 16 + lc;
      u16x16 at;
      at.lo = *(const v8u*)&lA[ar][akb];        // K = akb..akb+7
      at.hi = *(const v8u*)&lA[ar][akb + 8];    // K = akb+8..akb+15
      const v16bf af = __builtin_bit_cast(v16bf, at);
      #pragma unroll
      for (int sn = 0; sn < 2; ++sn) {
        const int bc = waveN * 32 + sn * 16 + lc;
        u16x16 bt;
        bt.lo = *(const v8u*)&lBt[bc][hiL * 16];      // K = hiL*16 .. +7
        bt.hi = *(const v8u*)&lBt[bc][hiL * 16 + 8];  // K = hiL*16+8 .. +15
        const v16bf bf = __builtin_bit_cast(v16bf, bt);
        acc[sm][sn] = __builtin_amdgcn_wmma_f32_16x16x32_bf16(
            false, af, false, bf, (short)0, acc[sm][sn], false, false);
      }
    }
    __syncthreads();
  }

  // --- epilogue: bias + activation + store
  #pragma unroll
  for (int sm = 0; sm < 2; ++sm) {
    #pragma unroll
    for (int sn = 0; sn < 2; ++sn) {
      const int col = bn + waveN * 32 + sn * 16 + lc;
      const float bv = bias[col];
      #pragma unroll
      for (int i = 0; i < 8; ++i) {
        const int row = bm + waveM * 32 + sm * 16 + i + hiL * 8;
        float v = acc[sm][sn][i] + bv;
        if (RELU) v = v > 0.0f ? v : 0.0f;
        if (OUT_F32)
          ((float*)Out)[(size_t)row * D_MODEL + col] = v;
        else
          ((unsigned short*)Out)[(size_t)row * D_MODEL + col] = f32_to_bf16(v);
      }
    }
  }
}

// ---------------------------------------------------------------------------
// Attention: one block per (b,n), one wave per head.
//   scores[16x16] = q[16x64] @ k^T[64x16] / 8   (2x WMMA bf16 k=32)
//   softmax over p (row spread across 16 lanes -> width-16 shuffles)
//   ctx[16x64]    = attn[16x32(pad)] @ v[32x16] (4x WMMA)
// ---------------------------------------------------------------------------
__global__ void __launch_bounds__(256)
attn_kernel(const unsigned short* __restrict__ Q,
            const unsigned short* __restrict__ Kb,
            const unsigned short* __restrict__ V,
            unsigned short* __restrict__ Ctx,
            float* __restrict__ AttnOut) {
  __shared__ float sc[NHEAD][16][17];

  const int b    = blockIdx.x >> 10;       // / N_NODES
  const int n    = blockIdx.x & (N_NODES - 1);
  const int h    = threadIdx.x >> 5;       // head == wave
  const int lane = threadIdx.x & 31;
  const int lc   = lane & 15;
  const int hiL  = lane >> 4;
  const int akb  = hiL * 8;

  // ---- scores = q @ k^T * (1/8)
  v8f s = {};
  #pragma unroll
  for (int kc = 0; kc < 2; ++kc) {
    v16u au = {};  // A: row = q-time (lc), K = d within 32-chunk
    v16u bu = {};  // B: col = p-time (lc), K = d within 32-chunk
    if (lc < TT) {
      const unsigned short* qp =
          Q + ((size_t)((b * TT + lc) * N_NODES + n)) * D_MODEL + h * D_HEAD + kc * 32;
      const unsigned short* kp =
          Kb + ((size_t)((b * TT + lc) * N_NODES + n)) * D_MODEL + h * D_HEAD + kc * 32;
      #pragma unroll
      for (int e = 0; e < 16; ++e) au[e] = qp[hiL * 16 + e];
      #pragma unroll
      for (int e = 0; e < 16; ++e) bu[e] = kp[hiL * 16 + e];
    }
    s = __builtin_amdgcn_wmma_f32_16x16x32_bf16(
        false, __builtin_bit_cast(v16bf, au), false, __builtin_bit_cast(v16bf, bu),
        (short)0, s, false, false);
  }

  // ---- softmax over p (columns); lane holds column p=lc, VGPR i holds row i+hiL*8
  float a[8];
  #pragma unroll
  for (int i = 0; i < 8; ++i)
    a[i] = (lc < TT) ? s[i] * 0.125f : -__builtin_inff();
  #pragma unroll
  for (int i = 0; i < 8; ++i) {
    float m = a[i];
    #pragma unroll
    for (int off = 8; off >= 1; off >>= 1) m = fmaxf(m, __shfl_xor(m, off, 16));
    const float e = __expf(a[i] - m);
    float sum = e;
    #pragma unroll
    for (int off = 8; off >= 1; off >>= 1) sum += __shfl_xor(sum, off, 16);
    a[i] = e / sum;
  }

  // ---- emit attn probs ([K*B, N, TQ, TP]) and stash tile in LDS for transpose
  #pragma unroll
  for (int i = 0; i < 8; ++i) {
    const int r = i + hiL * 8;
    if (r < TT && lc < TT)
      AttnOut[((((size_t)h * BATCH + b) * N_NODES + n) * TT + r) * TT + lc] = a[i];
    sc[h][r][lc] = a[i];
  }
  __syncthreads();

  // ---- ctx = attn @ v : A frag = attn (16x32, K=p, zero-padded p>=12)
  v16u au2 = {};
  #pragma unroll
  for (int e = 0; e < 16; ++e) {
    const int p = hiL * 16 + e;
    au2[e] = (p < TT) ? f32_to_bf16(sc[h][lc][p]) : (unsigned short)0;
  }
  const v16bf af2 = __builtin_bit_cast(v16bf, au2);

  #pragma unroll
  for (int j = 0; j < 4; ++j) {   // 16-wide chunks of d_head
    v16u bu = {};
    #pragma unroll
    for (int e = 0; e < 16; ++e) {
      const int p = hiL * 16 + e;
      if (p < TT)
        bu[e] = V[((size_t)((b * TT + p) * N_NODES + n)) * D_MODEL +
                  h * D_HEAD + j * 16 + lc];
    }
    v8f c = {};
    c = __builtin_amdgcn_wmma_f32_16x16x32_bf16(
        false, af2, false, __builtin_bit_cast(v16bf, bu), (short)0, c, false, false);
    #pragma unroll
    for (int i = 0; i < 8; ++i) {
      const int r = i + hiL * 8;
      if (r < TT)
        Ctx[((size_t)((b * TT + r) * N_NODES + n)) * D_MODEL +
            h * D_HEAD + j * 16 + lc] = f32_to_bf16(c[i]);
    }
  }
}

// ---------------------------------------------------------------------------
extern "C" void kernel_launch(void* const* d_in, const int* in_sizes, int n_in,
                              void* d_out, int out_size, void* d_ws, size_t ws_size,
                              hipStream_t stream) {
  (void)in_sizes; (void)n_in; (void)out_size; (void)ws_size;

  const float* X     = (const float*)d_in[0];
  const float* STE_P = (const float*)d_in[1];
  const float* STE_Q = (const float*)d_in[2];
  const float* Wq = (const float*)d_in[3];  const float* bq = (const float*)d_in[4];
  const float* Wk = (const float*)d_in[5];  const float* bk = (const float*)d_in[6];
  const float* Wv = (const float*)d_in[7];  const float* bv = (const float*)d_in[8];
  const float* W1 = (const float*)d_in[9];  const float* b1 = (const float*)d_in[10];
  const float* W2 = (const float*)d_in[11]; const float* b2 = (const float*)d_in[12];

  float* out  = (float*)d_out;                                   // [B,TQ,N,D]
  float* attn = out + (size_t)M_ROWS * D_MODEL;                  // [K*B,N,TQ,TP]

  // workspace layout (bf16 buffers)
  const size_t WSZ = (size_t)D_MODEL * D_MODEL;                  // 262144 elems
  const size_t BUF = (size_t)M_ROWS * D_MODEL;                   // 50331648 elems
  unsigned short* wW   = (unsigned short*)d_ws;                  // 5*WSZ (transposed)
  unsigned short* bufQ = wW + 5 * WSZ;                           // also reused as h
  unsigned short* bufK = bufQ + BUF;
  unsigned short* bufV = bufK + BUF;
  unsigned short* bufC = bufV + BUF;

  // 1) weights -> bf16, transposed Wt[n][k]
  cvt_weights<<<(int)(WSZ / 256), 256, 0, stream>>>(Wq, Wk, Wv, W1, W2, wW);

  // 2-4) QKV projections (f32 activations in, relu, bf16 out)
  dim3 gg(M_ROWS / 128, D_MODEL / 64);
  gemm512<true, true, false><<<gg, 256, 0, stream>>>(STE_Q, wW,           bq, bufQ);
  gemm512<true, true, false><<<gg, 256, 0, stream>>>(STE_P, wW + WSZ,     bk, bufK);
  gemm512<true, true, false><<<gg, 256, 0, stream>>>(X,     wW + 2 * WSZ, bv, bufV);

  // 5) attention (one block per (b,n), wave per head)
  attn_kernel<<<BATCH * N_NODES, 256, 0, stream>>>(bufQ, bufK, bufV, bufC, attn);

  // 6) h = relu(ctx @ W1 + b1)  (bf16 in via async-LDS; reuse bufQ for h)
  gemm512<false, true, false><<<gg, 256, 0, stream>>>(bufC, wW + 3 * WSZ, b1, bufQ);

  // 7) out = h @ W2 + b2  (f32 out to d_out)
  gemm512<false, false, true><<<gg, 256, 0, stream>>>(bufQ, wW + 4 * WSZ, b2, out);
}
